// Speller_11089605558566
// MI455X (gfx1250) — compile-verified, matrix operand order checked by reading, not run on hardware
//
#include <hip/hip_runtime.h>
#include <cstdint>
#include <cstddef>

typedef _Float16 h16;
typedef _Float16 v16h __attribute__((ext_vector_type(16)));
typedef float    v8f  __attribute__((ext_vector_type(8)));

#define DEV __device__ __forceinline__

namespace C {
constexpr int B = 32, T = 1024, HC = 512, V = 128, HS = 512, P = 256, MAXLEN = 256;
constexpr int DIN = V + HC;          // 640
constexpr int G4  = 4 * HS;          // 2048
constexpr int NEOS = 3, EOS = 1;
constexpr float EPS_SM = 1e-5f, EPS_LN = 1e-5f;
constexpr int GRID = 32, WGS = 256;  // persistent decoder: 32 WGs (== B), 8 wave32 waves
constexpr size_t YHAT = (size_t)B * MAXLEN * V;   // 1,048,576
constexpr size_t ATT  = YHAT + (size_t)B;         // lens after y_hats

constexpr size_t al(size_t x) { return (x + 255) & ~(size_t)255; }
constexpr size_t oHH   = 0;                                               // h f16 [B*T*HC]
constexpr size_t oNH   = al(oHH   + sizeof(h16)   * (size_t)B * T * HC);  // n=psi(h) f16 [B*T*P]
constexpr size_t oW0I  = al(oNH   + sizeof(h16)   * (size_t)B * T * P);
constexpr size_t oW0H  = al(oW0I  + sizeof(h16)   * (size_t)G4 * DIN);
constexpr size_t oW1I  = al(oW0H  + sizeof(h16)   * (size_t)G4 * HS);
constexpr size_t oW1H  = al(oW1I  + sizeof(h16)   * (size_t)G4 * HS);
constexpr size_t oPHI  = al(oW1H  + sizeof(h16)   * (size_t)G4 * HS);
constexpr size_t oPSI  = al(oPHI  + sizeof(h16)   * (size_t)P * HS);
constexpr size_t oFC1  = al(oPSI  + sizeof(h16)   * (size_t)P * HC);
constexpr size_t oFC2  = al(oFC1  + sizeof(h16)   * (size_t)V * (HS + HC));
constexpr size_t oXH   = al(oFC2  + sizeof(h16)   * (size_t)V * V);       // decoder input f16 [B,DIN]
constexpr size_t oC0   = al(oXH   + sizeof(h16)   * (size_t)B * DIN);     // cell state f32
constexpr size_t oC1   = al(oC0   + sizeof(float) * (size_t)B * HS);
constexpr size_t oH0H  = al(oC1   + sizeof(float) * (size_t)B * HS);      // h0 f16
constexpr size_t oH1H  = al(oH0H  + sizeof(h16)   * (size_t)B * HS);      // h1 f16
constexpr size_t oSH   = al(oH1H  + sizeof(h16)   * (size_t)B * HS);      // layernorm out f16
constexpr size_t oG    = al(oSH   + sizeof(h16)   * (size_t)B * HS);      // gates f32 [B,G4]
constexpr size_t oCTXH = al(oG    + sizeof(float) * (size_t)B * G4);      // context f16 [B,HC]
constexpr size_t oHDNH = al(oCTXH + sizeof(h16)   * (size_t)B * HC);      // fc1 out f16 [B,128]
constexpr size_t oLG   = al(oHDNH + sizeof(h16)   * (size_t)B * V);       // logits f32 [B,V]
constexpr size_t oBMAX = al(oLG   + sizeof(float) * (size_t)B * V);       // per-WG score max
constexpr size_t oLENS = al(oBMAX + sizeof(float) * (size_t)GRID);
constexpr size_t oBI   = al(oLENS + sizeof(int)   * (size_t)B);
constexpr size_t oBAR  = al(oBI   + sizeof(int)   * (size_t)NEOS * B);
} // namespace C

// ---------------- WMMA helpers (wave32, 16x16x32 f16 -> f32) ------------------
DEV v16h load_tile_k32(const h16* __restrict__ Pm, int ld, int row0, int k0) {
  const int l  = threadIdx.x & 31;
  const int r  = l & 15;
  const int kb = k0 + ((l >> 4) << 3);
  const h16* p = Pm + (size_t)(row0 + r) * ld + kb;
  v16h v;
#pragma unroll
  for (int i = 0; i < 8; ++i) { v[i] = p[i]; v[8 + i] = p[16 + i]; }
  return v;
}

DEV v8f wmma32(v16h a, v16h b, v8f c) {
  return __builtin_amdgcn_wmma_f32_16x16x32_f16(false, a, false, b, (short)0, c,
                                                false, false);
}

DEV void store_tile_f32(float* D, int ld, int m0, int n0, v8f acc) {
  const int l = threadIdx.x & 31, r = l & 15, hf = l >> 4;
#pragma unroll
  for (int j = 0; j < 8; ++j) D[(size_t)(m0 + hf * 8 + j) * ld + n0 + r] = acc[j];
}
DEV void store_tile_f16(h16* D, int ld, int m0, int n0, v8f acc) {
  const int l = threadIdx.x & 31, r = l & 15, hf = l >> 4;
#pragma unroll
  for (int j = 0; j < 8; ++j) D[(size_t)(m0 + hf * 8 + j) * ld + n0 + r] = (h16)acc[j];
}

DEV v8f gemm_tile(const h16* A, int lda, const h16* Wt, int ldw, int m0, int n0,
                  int K, v8f acc) {
  for (int k0 = 0; k0 < K; k0 += 32)
    acc = wmma32(load_tile_k32(A, lda, m0, k0), load_tile_k32(Wt, ldw, n0, k0), acc);
  return acc;
}

DEV float sigm(float x) { return 1.f / (1.f + __expf(-x)); }

// ---- async global->LDS staging: 16 rows x 512 f16 = 16KB, 64B/thread --------
DEV void stage_chunk(unsigned lds_base, const h16* hrow0, int c, int tid) {
  const char* ga = (const char*)(hrow0 + (size_t)c * 16 * C::HC) + tid * 64;
  unsigned l = lds_base + (unsigned)tid * 64u;
  asm volatile("global_load_async_to_lds_b128 %0, %1, off\n\t"
               "global_load_async_to_lds_b128 %0, %1, off offset:16\n\t"
               "global_load_async_to_lds_b128 %0, %1, off offset:32\n\t"
               "global_load_async_to_lds_b128 %0, %1, off offset:48"
               :: "v"(l), "v"(ga) : "memory");
}

// ---------------- grid-wide barrier (persistent kernel) -----------------------
DEV void grid_sync(unsigned* bar) {
  __syncthreads();
  if (threadIdx.x == 0) {
    __threadfence();
    unsigned* cnt = bar;
    unsigned* gen = bar + 1;
    unsigned g = __hip_atomic_load(gen, __ATOMIC_RELAXED, __HIP_MEMORY_SCOPE_AGENT);
    unsigned a = __hip_atomic_fetch_add(cnt, 1u, __ATOMIC_ACQ_REL, __HIP_MEMORY_SCOPE_AGENT);
    if (a == (unsigned)C::GRID - 1) {
      __hip_atomic_store(cnt, 0u, __ATOMIC_RELAXED, __HIP_MEMORY_SCOPE_AGENT);
      __hip_atomic_fetch_add(gen, 1u, __ATOMIC_ACQ_REL, __HIP_MEMORY_SCOPE_AGENT);
    } else {
      while (__hip_atomic_load(gen, __ATOMIC_ACQUIRE, __HIP_MEMORY_SCOPE_AGENT) == g)
        __builtin_amdgcn_s_sleep(1);
    }
    __threadfence();
  }
  __syncthreads();
}

DEV float block_reduce_sum(float v, float* red) {
  const int tid = threadIdx.x;
  red[tid] = v; __syncthreads();
  for (int s = C::WGS / 2; s > 0; s >>= 1) {
    if (tid < s) red[tid] += red[tid + s];
    __syncthreads();
  }
  float r = red[0]; __syncthreads();
  return r;
}
DEV float block_reduce_max(float v, float* red) {
  const int tid = threadIdx.x;
  red[tid] = v; __syncthreads();
  for (int s = C::WGS / 2; s > 0; s >>= 1) {
    if (tid < s) red[tid] = fmaxf(red[tid], red[tid + s]);
    __syncthreads();
  }
  float r = red[0]; __syncthreads();
  return r;
}

// ---------------- kernels -----------------------------------------------------

__global__ void prep_kernel(char* __restrict__ ws, const float* __restrict__ h,
                            const float* __restrict__ Wih0, const float* __restrict__ Whh0,
                            const float* __restrict__ Wih1, const float* __restrict__ Whh1,
                            const float* __restrict__ phi,  const float* __restrict__ psi,
                            const float* __restrict__ fc1,  const float* __restrict__ fc2) {
  using namespace C;
  const size_t i0 = (size_t)blockIdx.x * blockDim.x + threadIdx.x;
  const size_t np = (size_t)gridDim.x * blockDim.x;
  h16* hh  = (h16*)(ws + oHH);
  h16* w0i = (h16*)(ws + oW0I);
  h16* w0h = (h16*)(ws + oW0H);
  h16* w1i = (h16*)(ws + oW1I);
  h16* w1h = (h16*)(ws + oW1H);
  h16* ph  = (h16*)(ws + oPHI);
  h16* ps  = (h16*)(ws + oPSI);
  h16* f1  = (h16*)(ws + oFC1);
  h16* f2  = (h16*)(ws + oFC2);
  h16* xh  = (h16*)(ws + oXH);
  float* c0 = (float*)(ws + oC0);
  float* c1 = (float*)(ws + oC1);
  h16* h0h = (h16*)(ws + oH0H);
  h16* h1h = (h16*)(ws + oH1H);
  for (size_t i = i0; i < (size_t)B * T * HC; i += np) hh[i] = (h16)h[i];
  for (size_t i = i0; i < (size_t)G4 * DIN;   i += np) w0i[i] = (h16)Wih0[i];
  for (size_t i = i0; i < (size_t)G4 * HS;    i += np) {
    w0h[i] = (h16)Whh0[i]; w1i[i] = (h16)Wih1[i]; w1h[i] = (h16)Whh1[i];
  }
  for (size_t i = i0; i < (size_t)P * HS; i += np) { ph[i] = (h16)phi[i]; ps[i] = (h16)psi[i]; }
  for (size_t i = i0; i < (size_t)V * (HS + HC); i += np) f1[i] = (h16)fc1[i];
  for (size_t i = i0; i < (size_t)V * V; i += np) f2[i] = (h16)fc2[i];
  for (size_t i = i0; i < (size_t)B * DIN; i += np) {   // x0 = [onehot(SOS), h[:,0,:]]
    int b = (int)(i / DIN), j = (int)(i % DIN);
    float v = (j == 0) ? 1.f : (j < V ? 0.f : h[(size_t)b * T * HC + (j - V)]);
    xh[i] = (h16)v;
  }
  for (size_t i = i0; i < (size_t)B * HS; i += np) {
    c0[i] = 0.f; c1[i] = 0.f; h0h[i] = (h16)0.f; h1h[i] = (h16)0.f;
  }
  int* lens = (int*)(ws + oLENS);
  int* bi   = (int*)(ws + oBI);
  unsigned* bar = (unsigned*)(ws + oBAR);
  if (i0 < (size_t)B) lens[i0] = MAXLEN;
  if (i0 < (size_t)NEOS * B) bi[i0] = 0;
  if (i0 < 2) bar[i0] = 0u;
}

// n = h @ psi^T : [32768 x 256 x 512] f16 WMMA GEMM (one tile per wave)
__global__ void psi_kernel(char* __restrict__ ws) {
  using namespace C;
  const h16* hh  = (const h16*)(ws + oHH);
  const h16* psw = (const h16*)(ws + oPSI);
  h16* n = (h16*)(ws + oNH);
  const int wave = blockIdx.x * (WGS / 32) + (threadIdx.x >> 5);
  const int ntiles = (B * T / 16) * (P / 16);  // 32768
  if (wave >= ntiles) return;                  // wave-uniform
  const int mt = wave >> 4, nt = wave & 15;
  v8f acc = {};
  acc = gemm_tile(hh, HC, psw, HC, mt * 16, nt * 16, HC, acc);
  store_tile_f16(n, P, mt * 16, nt * 16, acc);
}

// Persistent decoder: 32 WGs x 256 threads, 256 steps, 8 grid barriers/step.
__global__ void decoder_kernel(char* __restrict__ ws, float* __restrict__ out,
                               const int* __restrict__ xlen,
                               const float* __restrict__ bih0, const float* __restrict__ bhh0,
                               const float* __restrict__ bih1, const float* __restrict__ bhh1,
                               const float* __restrict__ phib, const float* __restrict__ fc1b) {
  using namespace C;
  h16*   xh   = (h16*)(ws + oXH);
  h16*   hh   = (h16*)(ws + oHH);
  h16*   nh   = (h16*)(ws + oNH);
  h16*   w0i  = (h16*)(ws + oW0I);
  h16*   w0h  = (h16*)(ws + oW0H);
  h16*   w1i  = (h16*)(ws + oW1I);
  h16*   w1h  = (h16*)(ws + oW1H);
  h16*   phih = (h16*)(ws + oPHI);
  h16*   fc1h = (h16*)(ws + oFC1);
  h16*   fc2h = (h16*)(ws + oFC2);
  float* c0   = (float*)(ws + oC0);
  float* c1   = (float*)(ws + oC1);
  h16*   h0h  = (h16*)(ws + oH0H);
  h16*   h1h  = (h16*)(ws + oH1H);
  h16*   sh   = (h16*)(ws + oSH);
  float* g    = (float*)(ws + oG);
  h16*   ctxh = (h16*)(ws + oCTXH);
  h16*   hdnh = (h16*)(ws + oHDNH);
  float* lg   = (float*)(ws + oLG);
  float* bmax = (float*)(ws + oBMAX);
  int*   lens = (int*)(ws + oLENS);
  int*   bi   = (int*)(ws + oBI);
  unsigned* bar = (unsigned*)(ws + oBAR);

  __shared__ float redf[C::WGS];
  __shared__ int   redi[C::WGS];
  __shared__ float s_lds[C::HS];        // layernorm output (f32)
  __shared__ float m_lds[C::P];         // phi projection
  __shared__ float e_lds[C::T];         // scores
  __shared__ float a_lds[C::T];         // attention weights
  __shared__ h16   hbuf[2][16 * C::HC]; // double-buffered h chunks (2 x 16KB)

  const int tid = threadIdx.x;
  const int gw  = blockIdx.x * (WGS / 32) + (tid >> 5);   // 0..255 global wave
  const int b   = blockIdx.x;                             // per-row phases: row == block
  const unsigned lb0 = (unsigned)(uintptr_t)(&hbuf[0][0]);
  const unsigned lb1 = (unsigned)(uintptr_t)(&hbuf[1][0]);

  for (int st = 0; st < MAXLEN; ++st) {
    // ---- P1: gates0 = x @ Wih0^T + h0 @ Whh0^T  (256 WMMA tiles, 1/wave)
    {
      const int mt = gw & 1, nt = gw >> 1;  // 2 x 128
      v8f acc = {};
      acc = gemm_tile(xh, DIN, w0i, DIN, mt * 16, nt * 16, DIN, acc);
      acc = gemm_tile(h0h, HS, w0h, HS, mt * 16, nt * 16, HS, acc);
      store_tile_f32(g, G4, mt * 16, nt * 16, acc);
    }
    grid_sync(bar);
    // ---- P2: LSTM0 pointwise, row b
    {
      const float* gr = g + (size_t)b * G4;
      const size_t rb = (size_t)b * HS;
      for (int j = tid; j < HS; j += WGS) {
        float gi = gr[j]          + bih0[j]          + bhh0[j];
        float gf = gr[HS + j]     + bih0[HS + j]     + bhh0[HS + j];
        float gg = gr[2 * HS + j] + bih0[2 * HS + j] + bhh0[2 * HS + j];
        float go = gr[3 * HS + j] + bih0[3 * HS + j] + bhh0[3 * HS + j];
        float cn = sigm(gf) * c0[rb + j] + sigm(gi) * tanhf(gg);
        c0[rb + j] = cn;
        h0h[rb + j] = (h16)(sigm(go) * tanhf(cn));
      }
    }
    grid_sync(bar);
    // ---- P3: gates1 = h0 @ Wih1^T + h1 @ Whh1^T
    {
      const int mt = gw & 1, nt = gw >> 1;
      v8f acc = {};
      acc = gemm_tile(h0h, HS, w1i, HS, mt * 16, nt * 16, HS, acc);
      acc = gemm_tile(h1h, HS, w1h, HS, mt * 16, nt * 16, HS, acc);
      store_tile_f32(g, G4, mt * 16, nt * 16, acc);
    }
    grid_sync(bar);
    // ---- P4: row b fused: LSTM1 -> LayerNorm -> phi -> attention scores
    {
      const float* gr = g + (size_t)b * G4;
      const size_t rb = (size_t)b * HS;
      const int j0 = tid, j1 = tid + 256;
      float x0v, x1v;
      {
        float gi = gr[j0]          + bih1[j0]          + bhh1[j0];
        float gf = gr[HS + j0]     + bih1[HS + j0]     + bhh1[HS + j0];
        float gg = gr[2 * HS + j0] + bih1[2 * HS + j0] + bhh1[2 * HS + j0];
        float go = gr[3 * HS + j0] + bih1[3 * HS + j0] + bhh1[3 * HS + j0];
        float cn = sigm(gf) * c1[rb + j0] + sigm(gi) * tanhf(gg);
        c1[rb + j0] = cn; x0v = sigm(go) * tanhf(cn);
        h1h[rb + j0] = (h16)x0v;
      }
      {
        float gi = gr[j1]          + bih1[j1]          + bhh1[j1];
        float gf = gr[HS + j1]     + bih1[HS + j1]     + bhh1[HS + j1];
        float gg = gr[2 * HS + j1] + bih1[2 * HS + j1] + bhh1[2 * HS + j1];
        float go = gr[3 * HS + j1] + bih1[3 * HS + j1] + bhh1[3 * HS + j1];
        float cn = sigm(gf) * c1[rb + j1] + sigm(gi) * tanhf(gg);
        c1[rb + j1] = cn; x1v = sigm(go) * tanhf(cn);
        h1h[rb + j1] = (h16)x1v;
      }
      // LayerNorm (row in registers: 2 elems/thread)
      float mu  = block_reduce_sum(x0v + x1v, redf) * (1.f / HS);
      float d0 = x0v - mu, d1 = x1v - mu;
      float var = block_reduce_sum(d0 * d0 + d1 * d1, redf) * (1.f / HS);
      float inv = rsqrtf(var + EPS_LN);
      float s0 = d0 * inv, s1 = d1 * inv;
      s_lds[j0] = s0; s_lds[j1] = s1;
      sh[rb + j0] = (h16)s0; sh[rb + j1] = (h16)s1;
      __syncthreads();
      // m = s @ phi^T + phi_b (one output column per thread, s from LDS broadcast)
      {
        const h16* prow = phih + (size_t)tid * HS;
        float acc = phib[tid];
#pragma unroll 8
        for (int k = 0; k < HS; ++k) acc += s_lds[k] * (float)prow[k];
        m_lds[tid] = acc;
      }
      __syncthreads();
      // scores e[b,t] = <m, n[b,t]> with prefetch of upcoming n rows
      float lmax = -1e30f;
      for (int t = tid; t < T; t += WGS) {
        const h16* nrow = nh + ((size_t)b * T + t) * P;
        __builtin_prefetch(nrow + (size_t)WGS * P, 0, 1);
        float s = 0.f;
#pragma unroll 8
        for (int p = 0; p < P; ++p) s += m_lds[p] * (float)nrow[p];
        e_lds[t] = s;
        lmax = fmaxf(lmax, s);
      }
      float wmax = block_reduce_max(lmax, redf);
      if (tid == 0) bmax[b] = wmax;
    }
    grid_sync(bar);  // need global max across all rows
    // ---- P5: row b fused: masked softmax (global max shift) -> context
    {
      float gmaxv = -1e30f;
      for (int i = 0; i < GRID; ++i) gmaxv = fmaxf(gmaxv, bmax[i]);
      const int len = xlen[b];
      float lsum = 0.f;
      for (int t = tid; t < T; t += WGS) {
        float ex = (t < len) ? __expf(e_lds[t] - gmaxv) : 0.f;
        a_lds[t] = ex; lsum += ex;
      }
      float inv = 1.f / (block_reduce_sum(lsum, redf) + EPS_SM);
      for (int t = tid; t < T; t += WGS) {
        float aa = a_lds[t] * inv;
        a_lds[t] = aa;
        out[ATT + ((size_t)b * MAXLEN + st) * T + t] = aa;
      }
      __syncthreads();
      // context: double-buffered async global->LDS staging of h (16-row chunks)
      const h16* hrow0 = hh + (size_t)b * T * HC;
      float s0 = 0.f, s1 = 0.f;
      stage_chunk(lb0, hrow0, 0, tid);
      int cur = 0;
      for (int c = 0; c < T / 16; ++c) {
        if (c + 1 < T / 16) {
          stage_chunk(cur ? lb0 : lb1, hrow0, c + 1, tid);
          asm volatile("s_wait_asynccnt 4" ::: "memory");   // chunk c landed
        } else {
          asm volatile("s_wait_asynccnt 0" ::: "memory");
        }
        __syncthreads();
        const h16* hb = hbuf[cur];
        const float* ap = a_lds + c * 16;
#pragma unroll
        for (int t = 0; t < 16; ++t) {
          float aa = ap[t];
          s0 += aa * (float)hb[t * HC + tid];
          s1 += aa * (float)hb[t * HC + 256 + tid];
        }
        __syncthreads();   // done reading buf `cur` before it is restaged
        cur ^= 1;
      }
      ctxh[(size_t)b * HC + tid]       = (h16)s0;
      ctxh[(size_t)b * HC + 256 + tid] = (h16)s1;
      xh[(size_t)b * DIN + V + tid]       = (h16)s0;  // next input, ctx part
      xh[(size_t)b * DIN + V + 256 + tid] = (h16)s1;
    }
    grid_sync(bar);
    // ---- P6: fc1 on concat([s, c]) (16 WMMA tiles, waves 0..15)
    if (gw < 16) {
      const int mt = gw & 1, nt = gw >> 1;  // 2 x 8
      v8f acc = {};
      for (int k0 = 0; k0 < HS; k0 += 32)
        acc = wmma32(load_tile_k32(sh, HS, mt * 16, k0),
                     load_tile_k32(fc1h, HS + HC, nt * 16, k0), acc);
      for (int k0 = 0; k0 < HC; k0 += 32)
        acc = wmma32(load_tile_k32(ctxh, HC, mt * 16, k0),
                     load_tile_k32(fc1h, HS + HC, nt * 16, HS + k0), acc);
      const int l = tid & 31, r = l & 15, hf = l >> 4;
      const int col = nt * 16 + r;
#pragma unroll
      for (int j = 0; j < 8; ++j)
        hdnh[(size_t)(mt * 16 + hf * 8 + j) * V + col] = (h16)(acc[j] + fc1b[col]);
    }
    grid_sync(bar);
    // ---- P7: fc2 logits (16 WMMA tiles, waves 0..15)
    if (gw < 16) {
      const int mt = gw & 1, nt = gw >> 1;
      v8f acc = {};
      acc = gemm_tile(hdnh, V, fc2h, V, mt * 16, nt * 16, V, acc);
      store_tile_f32(lg, V, mt * 16, nt * 16, acc);
    }
    grid_sync(bar);
    // ---- P8: row b: output softmax + argmax + EOS bookkeeping + next x (y part)
    {
      float v = (tid < V) ? lg[(size_t)b * V + tid] : -1e30f;
      float mx = block_reduce_max(v, redf);
      float ex = (tid < V) ? __expf(v - mx) : 0.f;
      float sm = block_reduce_sum(ex, redf);
      float y = ex / sm;
      if (tid < V) {
        out[((size_t)b * MAXLEN + st) * V + tid] = y;
        xh[(size_t)b * DIN + tid] = (h16)y;
      }
      redf[tid] = (tid < V) ? y : -1e30f;
      redi[tid] = tid;
      __syncthreads();
      for (int s = WGS / 2; s > 0; s >>= 1) {
        if (tid < s) {
          float ov = redf[tid + s]; int oi = redi[tid + s];
          if (ov > redf[tid] || (ov == redf[tid] && oi < redi[tid])) {
            redf[tid] = ov; redi[tid] = oi;
          }
        }
        __syncthreads();
      }
      if (tid == 0) {
        bi[(st % NEOS) * B + b] = (redi[0] == EOS) ? 1 : 0;
        int done = bi[0 * B + b] & bi[1 * B + b] & bi[2 * B + b];
        if (done && lens[b] > st) lens[b] = st + 1;
      }
    }
    grid_sync(bar);
  }
  if (blockIdx.x == 0 && tid < B) ((int*)out)[YHAT + tid] = lens[tid];
}

extern "C" void kernel_launch(void* const* d_in, const int* in_sizes, int n_in,
                              void* d_out, int out_size, void* d_ws, size_t ws_size,
                              hipStream_t stream) {
  (void)in_sizes; (void)n_in; (void)out_size; (void)ws_size;
  char* ws = (char*)d_ws;
  const float* h    = (const float*)d_in[0];
  const int*   xlen = (const int*)d_in[1];
  const float* Wih0 = (const float*)d_in[2];
  const float* Whh0 = (const float*)d_in[3];
  const float* bih0 = (const float*)d_in[4];
  const float* bhh0 = (const float*)d_in[5];
  const float* Wih1 = (const float*)d_in[6];
  const float* Whh1 = (const float*)d_in[7];
  const float* bih1 = (const float*)d_in[8];
  const float* bhh1 = (const float*)d_in[9];
  const float* phiw = (const float*)d_in[10];
  const float* phib = (const float*)d_in[11];
  const float* psiw = (const float*)d_in[12];
  const float* fc1w = (const float*)d_in[13];
  const float* fc1b = (const float*)d_in[14];
  const float* fc2w = (const float*)d_in[15];

  prep_kernel<<<2048, 256, 0, stream>>>(ws, h, Wih0, Whh0, Wih1, Whh1,
                                        phiw, psiw, fc1w, fc2w);
  psi_kernel<<<4096, 256, 0, stream>>>(ws);
  decoder_kernel<<<C::GRID, C::WGS, 0, stream>>>(ws, (float*)d_out, xlen,
                                                 bih0, bhh0, bih1, bhh1, phib, fc1b);
}
